// ConstrainedLinear_15582141350319
// MI455X (gfx1250) — compile-verified, matrix operand order checked by reading, not run on hardware
//
#include <hip/hip_runtime.h>
#include <hip/hip_bf16.h>

typedef __attribute__((ext_vector_type(16))) __bf16 v16bf;
typedef __attribute__((ext_vector_type(8)))  float  v8f;
typedef __attribute__((ext_vector_type(4)))  float  v4f;
typedef __attribute__((ext_vector_type(4)))  unsigned int v4u;
typedef __attribute__((ext_vector_type(2)))  unsigned int v2u;

#define BM  128
#define BN  128   // fallback kernel block-N
#define BNA 256   // async kernel block-N (64x64 per wave)
#define BK  32
#define LDK 40    // padded LDS row (elements) -> 80B rows, conflict-free b128 frag loads
#define TPB 256

__device__ __forceinline__ unsigned short f32_to_bf16(float f) {
    unsigned u = __float_as_uint(f);
    unsigned r = (u + 0x7FFFu + ((u >> 16) & 1u)) >> 16;   // round-to-nearest-even
    return (unsigned short)r;
}

union FragAB { v16bf v; v4f f4[2]; };

// ---------------------------------------------------------------------------
// One-shot fp32 -> packed bf16 conversion (8 elements / thread, 16B stores)
// ---------------------------------------------------------------------------
__global__ __launch_bounds__(TPB)
void ConstrainedLinear_cvt_bf16(const float* __restrict__ src,
                                unsigned short* __restrict__ dst,
                                long long n8) {
    long long i = (long long)blockIdx.x * TPB + threadIdx.x;
    if (i >= n8) return;
    const v4f* s = (const v4f*)src + i * 2;
    v4f a = s[0], b = s[1];
    v4u o;
    o.x = (unsigned)f32_to_bf16(a.x) | ((unsigned)f32_to_bf16(a.y) << 16);
    o.y = (unsigned)f32_to_bf16(a.z) | ((unsigned)f32_to_bf16(a.w) << 16);
    o.z = (unsigned)f32_to_bf16(b.x) | ((unsigned)f32_to_bf16(b.y) << 16);
    o.w = (unsigned)f32_to_bf16(b.z) | ((unsigned)f32_to_bf16(b.w) << 16);
    ((v4u*)dst)[i] = o;
}

// ---------------------------------------------------------------------------
// Async global->LDS helpers (ASYNCcnt-tracked CDNA5 path)
// ---------------------------------------------------------------------------
__device__ __forceinline__ void async_b128(unsigned lds_off, unsigned voff,
                                           unsigned long long sbase) {
    asm volatile("global_load_async_to_lds_b128 %0, %1, %2"
                 :: "v"(lds_off), "v"(voff), "s"(sbase) : "memory");
}
__device__ __forceinline__ void wait_async0() {
    asm volatile("s_wait_asynccnt 0x0" ::: "memory");
}

// ---------------------------------------------------------------------------
// Main GEMM: bf16 in global, async tile copies, double-buffered LDS,
// 128x256 block tile, 64x64 per wave (16 WMMA : 16 ds_load per k-step).
// Requires M%BM==0, N%BNA==0, K%BK==0.
// ---------------------------------------------------------------------------
__global__ __launch_bounds__(TPB)
void ConstrainedLinear_gemm_async(const unsigned short* __restrict__ xb,
                                  const unsigned short* __restrict__ wb,
                                  const float* __restrict__ bias,
                                  float* __restrict__ logits,
                                  int M, int N, int K) {
    const int tid  = threadIdx.x;
    const int col0 = blockIdx.x * BNA;
    const int row0 = blockIdx.y * BM;

    __shared__ __align__(16) unsigned short sA[2 * BM  * LDK];   // 20.0 KB
    __shared__ __align__(16) unsigned short sB[2 * BNA * LDK];   // 40.0 KB

    const int wid    = tid >> 5;
    const int lane   = tid & 31;
    const int wm     = wid >> 2;          // 0..1  : 64-row strip
    const int wn     = wid & 3;           // 0..3  : 64-col strip
    const int lane16 = lane & 15;
    const int khalf  = (lane >> 4) << 3;  // ISA 16-bit A/B lane layout

    // async geometry: 16B chunks; A tile = 512 chunks (2 instr/wave),
    // B tile = 1024 chunks (4 instr/wave)
    const int aChunk0 = wid * 64 + lane;
    const int aChunk1 = aChunk0 + 32;
    const unsigned ar0 = aChunk0 >> 2, ac0 = (aChunk0 & 3) << 4;
    const unsigned ar1 = aChunk1 >> 2, ac1 = (aChunk1 & 3) << 4;
    const unsigned K2  = (unsigned)(K * 2);
    const unsigned aVoff0 = ar0 * K2 + ac0, aRel0 = ar0 * (LDK * 2) + ac0;
    const unsigned aVoff1 = ar1 * K2 + ac1, aRel1 = ar1 * (LDK * 2) + ac1;

    unsigned bVoff[4], bRel[4];
#pragma unroll
    for (int c = 0; c < 4; ++c) {
        int chunk = wid * 128 + c * 32 + lane;
        unsigned r = chunk >> 2, cb = (chunk & 3) << 4;
        bVoff[c] = r * K2 + cb;
        bRel[c]  = r * (LDK * 2) + cb;
    }

    const unsigned ldsA = (unsigned)(uintptr_t)&sA[0];
    const unsigned ldsB = (unsigned)(uintptr_t)&sB[0];
    const unsigned long long gA0 =
        (unsigned long long)(uintptr_t)xb + (unsigned long long)row0 * K * 2ull;
    const unsigned long long gB0 =
        (unsigned long long)(uintptr_t)wb + (unsigned long long)col0 * K * 2ull;

    v8f acc[4][4];
#pragma unroll
    for (int i = 0; i < 4; ++i)
#pragma unroll
        for (int j = 0; j < 4; ++j) {
            v8f z = {0.f,0.f,0.f,0.f,0.f,0.f,0.f,0.f};
            acc[i][j] = z;
        }

    const int KT = K / BK;

    // prologue: tile 0 -> buffer 0
    async_b128(ldsA + aRel0, aVoff0, gA0);
    async_b128(ldsA + aRel1, aVoff1, gA0);
#pragma unroll
    for (int c = 0; c < 4; ++c) async_b128(ldsB + bRel[c], bVoff[c], gB0);

    for (int kt = 0; kt < KT; ++kt) {
        wait_async0();        // my share of tile kt is in LDS
        __syncthreads();      // all shares in; all waves done with kt-1

        if (kt + 1 < KT) {    // overlap tile kt+1 copies with WMMAs on tile kt
            const unsigned ab = ldsA + (unsigned)(((kt + 1) & 1) * BM  * LDK * 2);
            const unsigned bb = ldsB + (unsigned)(((kt + 1) & 1) * BNA * LDK * 2);
            const unsigned long long gak = gA0 + (unsigned long long)(kt + 1) * (BK * 2);
            const unsigned long long gbk = gB0 + (unsigned long long)(kt + 1) * (BK * 2);
            async_b128(ab + aRel0, aVoff0, gak);
            async_b128(ab + aRel1, aVoff1, gak);
#pragma unroll
            for (int c = 0; c < 4; ++c) async_b128(bb + bRel[c], bVoff[c], gbk);
        }

        const unsigned short* sAb = sA + (kt & 1) * (BM  * LDK);
        const unsigned short* sBb = sB + (kt & 1) * (BNA * LDK);

        FragAB b[4];
#pragma unroll
        for (int j = 0; j < 4; ++j) {
            int base = (wn * 64 + j * 16 + lane16) * LDK + khalf;
            b[j].f4[0] = *(const v4f*)&sBb[base];
            b[j].f4[1] = *(const v4f*)&sBb[base + 16];
        }
#pragma unroll
        for (int i = 0; i < 4; ++i) {
            FragAB a;
            int base = (wm * 64 + i * 16 + lane16) * LDK + khalf;
            a.f4[0] = *(const v4f*)&sAb[base];
            a.f4[1] = *(const v4f*)&sAb[base + 16];
#pragma unroll
            for (int j = 0; j < 4; ++j)
                acc[i][j] = __builtin_amdgcn_wmma_f32_16x16x32_bf16(
                    false, a.v, false, b[j].v, (short)0, acc[i][j], false, false);
        }
    }

#pragma unroll
    for (int j = 0; j < 4; ++j) {
        int cg = col0 + wn * 64 + j * 16 + lane16;
        float bv = bias[cg];
#pragma unroll
        for (int i = 0; i < 4; ++i) {
            int rbase = row0 + wm * 64 + i * 16 + ((lane >> 4) << 3);
#pragma unroll
            for (int p = 0; p < 8; ++p)
                logits[(size_t)(rbase + p) * N + cg] = acc[i][j][p] + bv;
        }
    }
}

// ---------------------------------------------------------------------------
// Fallback GEMM (inline fp32->bf16), used only if d_ws too small / odd dims.
// ---------------------------------------------------------------------------
__global__ __launch_bounds__(TPB)
void ConstrainedLinear_gemm_inline(const float* __restrict__ x,
                                   const float* __restrict__ w,
                                   const float* __restrict__ bias,
                                   float* __restrict__ logits,
                                   int M, int N, int K) {
    const int tid  = threadIdx.x;
    const int col0 = blockIdx.x * BN;
    const int row0 = blockIdx.y * BM;

    __shared__ __align__(16) unsigned short sA[BM * LDK];
    __shared__ __align__(16) unsigned short sB[BN * LDK];

    const int wid    = tid >> 5;
    const int lane   = tid & 31;
    const int wm     = wid >> 2;
    const int wn     = wid & 3;
    const int lane16 = lane & 15;
    const int khalf  = (lane >> 4) << 3;

    v8f acc[4][2];
#pragma unroll
    for (int i = 0; i < 4; ++i)
#pragma unroll
        for (int j = 0; j < 2; ++j) {
            v8f z = {0.f,0.f,0.f,0.f,0.f,0.f,0.f,0.f};
            acc[i][j] = z;
        }

    v4f ra[4], rb[4];
    auto loadG = [&](int k0) {
#pragma unroll
        for (int c = 0; c < 4; ++c) {
            int chunk = tid + c * TPB;
            int r     = chunk >> 3;
            int c4    = (chunk & 7) << 2;
            int rA = row0 + r; if (rA > M - 1) rA = M - 1;
            int rB = col0 + r; if (rB > N - 1) rB = N - 1;
            ra[c] = *(const v4f*)(x + (size_t)rA * K + k0 + c4);
            rb[c] = *(const v4f*)(w + (size_t)rB * K + k0 + c4);
        }
    };
    auto storeLDS = [&]() {
#pragma unroll
        for (int c = 0; c < 4; ++c) {
            int chunk = tid + c * TPB;
            int r     = chunk >> 3;
            int c4    = (chunk & 7) << 2;
            v2u pa, pb;
            pa.x = (unsigned)f32_to_bf16(ra[c].x) | ((unsigned)f32_to_bf16(ra[c].y) << 16);
            pa.y = (unsigned)f32_to_bf16(ra[c].z) | ((unsigned)f32_to_bf16(ra[c].w) << 16);
            pb.x = (unsigned)f32_to_bf16(rb[c].x) | ((unsigned)f32_to_bf16(rb[c].y) << 16);
            pb.y = (unsigned)f32_to_bf16(rb[c].z) | ((unsigned)f32_to_bf16(rb[c].w) << 16);
            *(v2u*)&sA[r * LDK + c4] = pa;
            *(v2u*)&sB[r * LDK + c4] = pb;
        }
    };

    const int KT = K / BK;
    loadG(0);
    for (int kt = 0; kt < KT; ++kt) {
        __syncthreads();
        storeLDS();
        __syncthreads();
        if (kt + 1 < KT) loadG((kt + 1) * BK);

        FragAB a[4], b[2];
#pragma unroll
        for (int i = 0; i < 4; ++i) {
            int base = (wm * 64 + i * 16 + lane16) * LDK + khalf;
            a[i].f4[0] = *(const v4f*)&sA[base];
            a[i].f4[1] = *(const v4f*)&sA[base + 16];
        }
#pragma unroll
        for (int j = 0; j < 2; ++j) {
            int base = (wn * 32 + j * 16 + lane16) * LDK + khalf;
            b[j].f4[0] = *(const v4f*)&sB[base];
            b[j].f4[1] = *(const v4f*)&sB[base + 16];
        }
#pragma unroll
        for (int i = 0; i < 4; ++i)
#pragma unroll
            for (int j = 0; j < 2; ++j)
                acc[i][j] = __builtin_amdgcn_wmma_f32_16x16x32_bf16(
                    false, a[i].v, false, b[j].v, (short)0, acc[i][j], false, false);
    }

#pragma unroll
    for (int j = 0; j < 2; ++j) {
        int cg = col0 + wn * 32 + j * 16 + lane16;
        float bv = (cg < N) ? bias[cg] : 0.f;
#pragma unroll
        for (int i = 0; i < 4; ++i) {
            int rbase = row0 + wm * 64 + i * 16 + ((lane >> 4) << 3);
#pragma unroll
            for (int p = 0; p < 8; ++p) {
                int rg = rbase + p;
                if (rg < M && cg < N)
                    logits[(size_t)rg * N + cg] = acc[i][j][p] + bv;
            }
        }
    }
}

// ---------------------------------------------------------------------------
// Per-row trie gather + -inf masking (block owns its row; barrier-ordered).
// ---------------------------------------------------------------------------
__global__ __launch_bounds__(TPB)
void ConstrainedLinear_trie_mask(const int* __restrict__ cur_node,
                                 const int* __restrict__ offsets,
                                 const int* __restrict__ tokens,
                                 const int* __restrict__ childs,
                                 float* __restrict__ out_next,
                                 float* __restrict__ out_vidx,
                                 float* __restrict__ logits,
                                 int V, int E) {
    const int b   = blockIdx.x;
    const int tid = threadIdx.x;

    __shared__ int   s_tok[64];
    __shared__ float s_val[64];
    __shared__ int   s_ok[64];

    float* row = logits + (size_t)b * V;

    if (tid < 64) {
        int node  = cur_node[b];
        int start = offsets[node];
        int deg   = offsets[node + 1] - start;
        int idx   = start + tid;
        if (idx > E - 1) idx = E - 1;
        int ok = tid < deg;
        int t  = tokens[idx];
        int c  = childs[idx];
        out_next[(size_t)b * 64 + tid] = ok ? (float)c : -1.0f;
        out_vidx[(size_t)b * 64 + tid] = ok ? (float)t : -1.0f;
        s_tok[tid] = t;
        s_ok[tid]  = ok;
        s_val[tid] = ok ? row[t] : 0.0f;
    }
    __syncthreads();

    const float NEG_INF = -__builtin_huge_valf();
    v4f ninf = {NEG_INF, NEG_INF, NEG_INF, NEG_INF};
    int V4 = V >> 2;
    for (int j = tid; j < V4; j += blockDim.x) ((v4f*)row)[j] = ninf;
    for (int j = (V4 << 2) + tid; j < V; j += blockDim.x) row[j] = NEG_INF;
    __syncthreads();

    if (tid < 64 && s_ok[tid]) row[s_tok[tid]] = s_val[tid];
}

// ---------------------------------------------------------------------------
extern "C" void kernel_launch(void* const* d_in, const int* in_sizes, int n_in,
                              void* d_out, int out_size, void* d_ws, size_t ws_size,
                              hipStream_t stream) {
    const float* x      = (const float*)d_in[0];
    const float* w      = (const float*)d_in[1];
    const float* bias   = (const float*)d_in[2];
    const int*   cur    = (const int*)d_in[3];
    const int*   offs   = (const int*)d_in[4];
    const int*   toks   = (const int*)d_in[5];
    const int*   childs = (const int*)d_in[6];

    const int V = in_sizes[2];
    const int D = in_sizes[1] / V;
    const int B = in_sizes[0] / D;
    const int E = in_sizes[5];

    float* out_next = (float*)d_out;
    float* out_vidx = out_next + (size_t)B * 64;
    float* logits   = out_vidx + (size_t)B * 64;

    const long long nW = (long long)V * D;
    const long long nX = (long long)B * D;
    const size_t needW = (size_t)nW * 2, needX = (size_t)nX * 2;

    const bool pre = (ws_size >= needW + needX) &&
                     (B % BM == 0) && (V % BNA == 0) && (D % BK == 0) &&
                     (nW % 8 == 0) && (nX % 8 == 0);

    if (pre) {
        unsigned short* wb = (unsigned short*)d_ws;
        unsigned short* xb = wb + nW;
        long long w8 = nW / 8, x8 = nX / 8;
        ConstrainedLinear_cvt_bf16<<<(unsigned)((w8 + TPB - 1) / TPB), TPB, 0, stream>>>(w, wb, w8);
        ConstrainedLinear_cvt_bf16<<<(unsigned)((x8 + TPB - 1) / TPB), TPB, 0, stream>>>(x, xb, x8);
        dim3 grid(V / BNA, B / BM);
        ConstrainedLinear_gemm_async<<<grid, TPB, 0, stream>>>(xb, wb, bias, logits, B, V, D);
    } else {
        dim3 g2((V + BN - 1) / BN, (B + BM - 1) / BM);
        ConstrainedLinear_gemm_inline<<<g2, TPB, 0, stream>>>(x, w, bias, logits, B, V, D);
    }
    ConstrainedLinear_trie_mask<<<B, TPB, 0, stream>>>(cur, offs, toks, childs,
                                                       out_next, out_vidx, logits, V, E);
}
